// Damping_82884278878217
// MI455X (gfx1250) — matrix operand order repaired and checked by reading, not compiled, not run on hardware
//
#include <hip/hip_runtime.h>

typedef __attribute__((ext_vector_type(16))) _Float16 v16h;
typedef __attribute__((ext_vector_type(8)))  float    v8f;

#define WAVES 8
#define H2S   33   // padded LDS row stride (floats) to dodge bank conflicts

__device__ __forceinline__ float tanh_hw(float x) {
#if __has_builtin(__builtin_amdgcn_tanhf)
  return __builtin_amdgcn_tanhf(x);
#else
  float r;
  asm volatile("v_tanh_f32 %0, %1\n\ts_nop 1" : "=v"(r) : "v"(x));
  return r;
#endif
}

__global__ __launch_bounds__(256) void damping_kernel(
    const float* __restrict__ X,
    const float* __restrict__ wd1, const float* __restrict__ wd2, const float* __restrict__ wd3,
    const float* __restrict__ wo1, const float* __restrict__ wo2, const float* __restrict__ wo3,
    const float* __restrict__ bd1, const float* __restrict__ bd2, const float* __restrict__ bd3,
    const float* __restrict__ bo1, const float* __restrict__ bo2, const float* __restrict__ bo3,
    float* __restrict__ out, int ntiles)
{
  __shared__ float s_wd1[64], s_bd1[32], s_wd2[1024], s_bd2[32], s_wd3[64], s_bd3[2];
  __shared__ float s_wo1[64], s_bo1[32], s_wo2[1024], s_bo2[32], s_wo3[32], s_bo3[2];
  __shared__ float s_x  [WAVES][32];
  __shared__ float s_h2d[WAVES][16 * H2S];
  __shared__ float s_h2o[WAVES][16 * H2S];

  const int tid = threadIdx.x;

  // Stage all weights/biases in LDS once per block (coalesced).
  for (int i = tid; i < 1024; i += 256) { s_wd2[i] = wd2[i]; s_wo2[i] = wo2[i]; }
  for (int i = tid; i < 64;   i += 256) { s_wd1[i] = wd1[i]; s_wo1[i] = wo1[i]; s_wd3[i] = wd3[i]; }
  for (int i = tid; i < 32;   i += 256) {
    s_bd1[i] = bd1[i]; s_bd2[i] = bd2[i]; s_bo1[i] = bo1[i]; s_bo2[i] = bo2[i]; s_wo3[i] = wo3[i];
  }
  if (tid < 2)  s_bd3[tid] = bd3[tid];
  if (tid == 0) s_bo3[0]   = bo3[0];
  __syncthreads();

  const int wave = tid >> 5;
  const int lane = tid & 31;
  const int m    = lane & 15;   // row (A) or column (B/D) index
  const int half = lane >> 4;   // which K-half this lane carries

  float* h2d = s_h2d[wave];
  float* h2o = s_h2o[wave];
  float* sx  = s_x[wave];

  // Weights-stationary WMMA B operands (32x16 f16 each):
  // lane holds column n = m; element e holds K = 16*half + e.
  v16h Bd0, Bd1, Bo0, Bo1;
#pragma unroll
  for (int e = 0; e < 16; ++e) {
    int k = 16 * half + e;
    Bd0[e] = (_Float16)s_wd2[k * 32 + m];
    Bd1[e] = (_Float16)s_wd2[k * 32 + 16 + m];
    Bo0[e] = (_Float16)s_wo2[k * 32 + m];
    Bo1[e] = (_Float16)s_wo2[k * 32 + 16 + m];
  }

  // Layer-2 bias folded into the C operand: every C/D element of a lane has N = m.
  v8f cd0, cd1, co0, co1;
#pragma unroll
  for (int v = 0; v < 8; ++v) {
    cd0[v] = s_bd2[m];      cd1[v] = s_bd2[16 + m];
    co0[v] = s_bo2[m];      co1[v] = s_bo2[16 + m];
  }

  const int gwave  = blockIdx.x * WAVES + wave;
  const int nwaves = gridDim.x * WAVES;

  for (int tile = gwave; tile < ntiles; tile += nwaves) {
    const int base = tile * 16;

    // 16 rows x 2 cols of x = 32 consecutive floats: one coalesced b32/lane.
    sx[lane] = X[base * 2 + lane];
    const float x0 = sx[2 * m];
    const float x1 = sx[2 * m + 1];

    // Layer 1 (K=2): VALU straight into WMMA A-operand lane layout.
    // element e: K = 16*half + e.
    v16h Ad, Ao;
#pragma unroll
    for (int e = 0; e < 16; ++e) {
      int k = 16 * half + e;
      float hd = tanh_hw(fmaf(x1, s_wd1[32 + k], fmaf(x0, s_wd1[k], s_bd1[k])));
      float ho = tanh_hw(fmaf(x1, s_wo1[32 + k], fmaf(x0, s_wo1[k], s_bo1[k])));
      Ad[e] = (_Float16)hd;
      Ao[e] = (_Float16)ho;
    }

    // Layer 2: [16x32] @ [32x32] -> 2 WMMAs per network, K=32 in one shot.
    v8f d0 = __builtin_amdgcn_wmma_f32_16x16x32_f16(false, Ad, false, Bd0, (short)0, cd0, false, false);
    v8f d1 = __builtin_amdgcn_wmma_f32_16x16x32_f16(false, Ad, false, Bd1, (short)0, cd1, false, false);
    v8f o0 = __builtin_amdgcn_wmma_f32_16x16x32_f16(false, Ao, false, Bo0, (short)0, co0, false, false);
    v8f o1 = __builtin_amdgcn_wmma_f32_16x16x32_f16(false, Ao, false, Bo1, (short)0, co1, false, false);

    // tanh (hardware TRANS) + scatter to LDS row-major for layer 3.
#pragma unroll
    for (int v = 0; v < 8; ++v) {
      int row = v + 8 * half;                       // D layout: M = v + 8*(lane/16)
      h2d[row * H2S + m]      = tanh_hw(d0[v]);
      h2d[row * H2S + 16 + m] = tanh_hw(d1[v]);
      h2o[row * H2S + m]      = tanh_hw(o0[v]);
      h2o[row * H2S + 16 + m] = tanh_hw(o1[v]);
    }

    // Layer 3: d3[16x2] (lane: row m, col half) and o3[16x1] (row m).
    float sd = 0.f, so = 0.f;
#pragma unroll
    for (int k = 0; k < 32; ++k) {
      sd = fmaf(h2d[m * H2S + k], s_wd3[k * 2 + half], sd);
      so = fmaf(h2o[m * H2S + k], s_wo3[k], so);
    }
    const float d3v = sd + s_bd3[half];
    const float c   = so + s_bo3[0];

    // Gather d3 col0/col1 for row m across the half-waves.
    const float a_raw = __shfl(d3v, m, 32);
    const float b_raw = __shfl(d3v, m + 16, 32);

    const float a  = (fmaxf(a_raw, 0.f) + 1.0e-3f) * x0;
    const float bb = (fmaxf(b_raw, 0.f) + 1.0e-3f) * x1;
    const float D0 = a * a * x0 + a * c * x1;
    const float D1 = a * c * x0 + (c * c + bb * bb) * x1;

    if (half == 0) {
      float2* o2p = (float2*)out;
      o2p[base + m] = make_float2(D0, D1);   // 16 lanes x 8B contiguous
    }
  }
}

extern "C" void kernel_launch(void* const* d_in, const int* in_sizes, int n_in,
                              void* d_out, int out_size, void* d_ws, size_t ws_size,
                              hipStream_t stream) {
  const float* X   = (const float*)d_in[0];
  const float* wd1 = (const float*)d_in[1];
  const float* wd2 = (const float*)d_in[2];
  const float* wd3 = (const float*)d_in[3];
  const float* wo1 = (const float*)d_in[4];
  const float* wo2 = (const float*)d_in[5];
  const float* wo3 = (const float*)d_in[6];
  const float* bd1 = (const float*)d_in[7];
  const float* bd2 = (const float*)d_in[8];
  const float* bd3 = (const float*)d_in[9];
  const float* bo1 = (const float*)d_in[10];
  const float* bo2 = (const float*)d_in[11];
  const float* bo3 = (const float*)d_in[12];
  float* out = (float*)d_out;

  const int n      = in_sizes[0] / 2;   // rows (B)
  const int ntiles = n / 16;            // B = 1,048,576 -> 65,536 tiles

  int blocks = 1024;                    // 8192 waves -> 8 tiles/wave
  int maxb   = (ntiles + WAVES - 1) / WAVES;
  if (blocks > maxb) blocks = maxb;
  if (blocks < 1) blocks = 1;

  damping_kernel<<<blocks, 256, 0, stream>>>(
      X, wd1, wd2, wd3, wo1, wo2, wo3,
      bd1, bd2, bd3, bo1, bo2, bo3, out, ntiles);
}
